// ReasonedCompressionLayer_29197187678611
// MI455X (gfx1250) — matrix operand order repaired and checked by reading, not compiled, not run on hardware
//
#include <hip/hip_runtime.h>
#include <cstdint>
#include <cstddef>

// ---------------------------------------------------------------------------
// CDNA5 (gfx1250) types / WMMA helpers
// ---------------------------------------------------------------------------
typedef __attribute__((ext_vector_type(16))) __bf16 v16bf;
typedef __attribute__((ext_vector_type(8)))  __bf16 v8bf;
typedef __attribute__((ext_vector_type(8)))  float  v8f;
typedef __attribute__((ext_vector_type(4)))  unsigned int v4u;
typedef __attribute__((ext_vector_type(8)))  int v8i;
typedef __attribute__((ext_vector_type(4)))  int v4i;

union V16U { v16bf v; v8bf h[2]; };

#define WMMA_BF16(a, b, c) \
  __builtin_amdgcn_wmma_f32_16x16x32_bf16(false, (a), false, (b), (short)0, (c), false, false)

// Build a 16-element bf16 fragment from two aligned 8-element LDS chunks.
static __device__ __forceinline__ v16bf ld_frag(const __bf16* p0, const __bf16* p1) {
  V16U u;
  u.h[0] = *(const v8bf*)p0;
  u.h[1] = *(const v8bf*)p1;
  return u.v;
}

// Low 32 bits of a generic pointer to a __shared__ object == LDS byte address
// (flat LDS aperture maps addr[31:0] directly to the workgroup LDS offset).
static __device__ __forceinline__ unsigned lds_addr32(const void* p) {
  return (unsigned)(unsigned long long)(uintptr_t)p;
}

// ---------------------------------------------------------------------------
// Tensor Data Mover: async 2D tile (bf16) global -> LDS, with LDS row padding
// done by the TDM pad engine. Tensor dims == tile dims (tiles always in-bounds).
//   pad_interval: 3 => pad after 16 DWORDs (64B rows), 4 => after 32 DWORDs (128B)
//   pad_amount  : 3 => insert 4 DWORDs (16B) of padding
// D# layout per CDNA5 ISA 8.3/8.4 (group0 128b, group1 256b).
// ---------------------------------------------------------------------------
static __device__ __forceinline__ void tdm_load_2d(const void* gaddr,
                                                   unsigned lds_byte_addr,
                                                   unsigned tile_d0, unsigned tile_d1,
                                                   unsigned stride_elems,
                                                   unsigned pad_interval, unsigned pad_amount)
{
  unsigned long long ga = (unsigned long long)(uintptr_t)gaddr;
  v4u g0;
  g0[0] = 1u;                                            // count=1, user D#
  g0[1] = lds_byte_addr;                                 // lds_addr
  g0[2] = (unsigned)(ga & 0xFFFFFFFFu);                  // global_addr[31:0]
  g0[3] = (unsigned)((ga >> 32) & 0x1FFFFFFu) | (2u << 30); // addr[56:32] | type=2
  v8i g1;
  g1[0] = (int)((1u << 16)                               // data_size = 2 bytes
              | (1u << 20)                               // pad_enable
              | (pad_interval << 22)
              | (pad_amount << 25));
  g1[1] = (int)((tile_d0 & 0xFFFFu) << 16);              // tensor_dim0[15:0]
  g1[2] = (int)(((tile_d0 >> 16) & 0xFFFFu) | ((tile_d1 & 0xFFFFu) << 16));
  g1[3] = (int)(((tile_d1 >> 16) & 0xFFFFu) | ((tile_d0 & 0xFFFFu) << 16)); // tile_dim0
  g1[4] = (int)(tile_d1 & 0xFFFFu);                      // tile_dim1 (tile_dim2=0)
  g1[5] = (int)stride_elems;                             // tensor_dim0_stride[31:0]
  g1[6] = 0;
  g1[7] = 0;
  v4i z4 = {0, 0, 0, 0};
#if defined(__clang_major__) && __clang_major__ >= 23
  v8i z8 = {0, 0, 0, 0, 0, 0, 0, 0};
  __builtin_amdgcn_tensor_load_to_lds(g0, g1, z4, z4, z8, 0);
#else
  __builtin_amdgcn_tensor_load_to_lds(g0, g1, z4, z4, 0);
#endif
}

// ---------------------------------------------------------------------------
// fp32 -> bf16 convert (grid-stride)
// ---------------------------------------------------------------------------
__global__ void cvt_f32_bf16(const float* __restrict__ s, __bf16* __restrict__ d, size_t n) {
  size_t i  = (size_t)blockIdx.x * blockDim.x + threadIdx.x;
  size_t st = (size_t)gridDim.x * blockDim.x;
  for (; i < n; i += st) d[i] = (__bf16)s[i];
}

// ---------------------------------------------------------------------------
// Generic WMMA GEMM: C[M,N] = A[M,K] * W[N,K]^T + bias  (A,W bf16; acc fp32)
// Block tile 128x128, K-step 32, TDM-staged double-buffered LDS.
// Requires M%128==0, N%128==0, K%32==0 (true for every GEMM in this model).
// ---------------------------------------------------------------------------
template <bool OUTBF>
__global__ __launch_bounds__(256) void gemm_bf16_wmma(
    const __bf16* __restrict__ A, const __bf16* __restrict__ W,
    const float* __restrict__ bias, void* __restrict__ Cout,
    int M, int N, int K)
{
  (void)M;
  constexpr int LD = 40;  // 32 data + 8 pad bf16 (TDM pad engine fills the gap)
  __shared__ __bf16 As[2][128 * LD];
  __shared__ __bf16 Bs[2][128 * LD];

  const int row0 = blockIdx.y * 128;
  const int col0 = blockIdx.x * 128;
  const int tid  = threadIdx.x;
  const int lane = tid & 31;
  const int wv   = tid >> 5;
  const int wm   = wv >> 1;   // 0..3
  const int wn   = wv & 1;    // 0..1
  const int g    = lane >> 4; // half-wave group
  const int mn   = lane & 15;
  const bool ldr = (tid < 32);  // TDM-issuing wave

  v8f acc[2][4] = {};

  // prologue: stage K-tile 0 into buffer 0
  if (ldr) {
    tdm_load_2d(&A[(size_t)row0 * K], lds_addr32(&As[0][0]), 32, 128, (unsigned)K, 3, 3);
    tdm_load_2d(&W[(size_t)col0 * K], lds_addr32(&Bs[0][0]), 32, 128, (unsigned)K, 3, 3);
    __builtin_amdgcn_s_wait_tensorcnt(0);
  }
  __syncthreads();

  int s = 0;
  for (int kk = 0; kk < K; kk += 32) {
    if (kk + 32 < K && ldr) {  // async-stage next K-tile into the other buffer
      tdm_load_2d(&A[(size_t)row0 * K + kk + 32], lds_addr32(&As[s ^ 1][0]),
                  32, 128, (unsigned)K, 3, 3);
      tdm_load_2d(&W[(size_t)col0 * K + kk + 32], lds_addr32(&Bs[s ^ 1][0]),
                  32, 128, (unsigned)K, 3, 3);
    }

    // A fragment (16x32): lane {g,mn} holds row mn, K = {g*8..+7, 16+g*8..+7}
    v16bf af[2];
#pragma unroll
    for (int i = 0; i < 2; i++) {
      const __bf16* p = &As[s][(wm * 32 + i * 16 + mn) * LD + g * 8];
      af[i] = ld_frag(p, p + 16);
    }
    // B fragment (32x16): lane {g,mn} holds col mn, K = {g*8..+7, 16+g*8..+7}
    v16bf bf[4];
#pragma unroll
    for (int j = 0; j < 4; j++) {
      const __bf16* p = &Bs[s][(wn * 64 + j * 16 + mn) * LD + g * 8];
      bf[j] = ld_frag(p, p + 16);
    }
#pragma unroll
    for (int i = 0; i < 2; i++)
#pragma unroll
      for (int j = 0; j < 4; j++)
        acc[i][j] = WMMA_BF16(af[i], bf[j], acc[i][j]);

    if (kk + 32 < K) {
      if (ldr) __builtin_amdgcn_s_wait_tensorcnt(0);
      __syncthreads();
      s ^= 1;
    }
  }

  // C layout: VGPR r -> row r + 8*g, col = mn
#pragma unroll
  for (int i = 0; i < 2; i++)
#pragma unroll
    for (int j = 0; j < 4; j++)
#pragma unroll
      for (int r = 0; r < 8; r++) {
        int row = row0 + wm * 32 + i * 16 + r + 8 * g;
        int col = col0 + wn * 64 + j * 16 + mn;
        float v = acc[i][j][r];
        if (bias) v += bias[col];
        if (OUTBF) ((__bf16*)Cout)[(size_t)row * N + col] = (__bf16)v;
        else       ((float*)Cout)[(size_t)row * N + col]  = v;
      }
}

// ---------------------------------------------------------------------------
// Flash attention (per-head), bf16 WMMA, online softmax.
// grid = (S/64, H, B), block = 128 (4 waves); wave w handles 16 query rows.
// Q/K staged by TDM; V transposed manually; row sums via P x ones WMMA.
// ---------------------------------------------------------------------------
__global__ __launch_bounds__(128) void flash_attn(const __bf16* __restrict__ qkv,
                                                  __bf16* __restrict__ o)
{
  constexpr int D = 1024, D3 = 3072, S = 1024;
  __shared__ __bf16 Qs[64 * 72];     // 64 queries x 64 dims (TDM pad -> pitch 72)
  __shared__ __bf16 Kt[32 * 72];     // 32 keys x 64 dims
  __shared__ __bf16 VtT[64 * 40];    // V transposed: [dim][key] (pitch 40)
  __shared__ __bf16 Ps[4 * 16 * 40]; // per-wave P staging (C-layout -> A-layout)

  const int qt = blockIdx.x, h = blockIdx.y, b = blockIdx.z;
  const int tid = threadIdx.x, lane = tid & 31, w = tid >> 5;
  const int g = lane >> 4, mn = lane & 15;
  const int q0 = qt * 64;
  const __bf16* base = qkv + (size_t)b * S * D3 + h * 64;
  const bool ldr = (tid < 32);

  // stage Q via TDM (64x64 tile, row stride D3, pad 16B per 128B row)
  if (ldr) {
    tdm_load_2d(base + (size_t)q0 * D3, lds_addr32(Qs), 64, 64, (unsigned)D3, 4, 3);
    __builtin_amdgcn_s_wait_tensorcnt(0);
  }
  __syncthreads();

  v16bf aq[2];
#pragma unroll
  for (int dh = 0; dh < 2; dh++) {
    const __bf16* p = &Qs[(w * 16 + mn) * 72 + dh * 32 + g * 8];
    aq[dh] = ld_frag(p, p + 16);
  }

  v16bf onesf;
#pragma unroll
  for (int i = 0; i < 16; i++) onesf[i] = (__bf16)1.0f;

  v8f accO[4] = {};
  float mrow[8], srow[8];
#pragma unroll
  for (int r = 0; r < 8; r++) { mrow[r] = -3.0e38f; srow[r] = 0.f; }
  __bf16* Pw = &Ps[w * 640];

  for (int kt = 0; kt < 32; kt++) {
    // stage K tile (TDM) and transposed V tile (manual scatter)
    if (ldr)
      tdm_load_2d(base + (size_t)(kt * 32) * D3 + 1024, lds_addr32(Kt),
                  64, 32, (unsigned)D3, 4, 3);
#pragma unroll
    for (int i = 0; i < 2; i++) {
      int c = tid + 128 * i;
      int r = c >> 3, cc = (c & 7) * 8;
      v8bf vv = *(const v8bf*)(base + (size_t)(kt * 32 + r) * D3 + 2048 + cc);
#pragma unroll
      for (int e = 0; e < 8; e++) VtT[(cc + e) * 40 + r] = vv[e];
      if (kt + 1 < 32)  // gfx1250 global_prefetch_b8 of next V tile
        __builtin_prefetch(base + (size_t)((kt + 1) * 32 + r) * D3 + 2048 + cc, 0, 1);
    }
    if (ldr) __builtin_amdgcn_s_wait_tensorcnt(0);
    __syncthreads();

    // S = Q * K^T (two 16x16 column chunks)
    v8f accS[2] = {};
#pragma unroll
    for (int nc = 0; nc < 2; nc++)
#pragma unroll
      for (int dh = 0; dh < 2; dh++) {
        const __bf16* p = &Kt[(nc * 16 + mn) * 72 + dh * 32 + g * 8];
        v16bf bk = ld_frag(p, p + 16);
        accS[nc] = WMMA_BF16(aq[dh], bk, accS[nc]);
      }

    // online softmax: per-row max via 16-lane shfl chain
    float corr[8], p0[8], p1[8];
#pragma unroll
    for (int r = 0; r < 8; r++) {
      float s0 = accS[0][r] * 0.125f;   // 1/sqrt(64)
      float s1 = accS[1][r] * 0.125f;
      float t = fmaxf(s0, s1);
#pragma unroll
      for (int x = 1; x < 16; x <<= 1) t = fmaxf(t, __shfl_xor(t, x, 32));
      float nm = fmaxf(mrow[r], t);
      corr[r] = __expf(mrow[r] - nm);
      mrow[r] = nm;
      p0[r] = __expf(s0 - nm);
      p1[r] = __expf(s1 - nm);
#pragma unroll
      for (int oc = 0; oc < 4; oc++) accO[oc][r] *= corr[r];
    }

    // C-layout P -> A-layout via per-wave LDS (DS in-order per wave)
    __builtin_amdgcn_wave_barrier();
#pragma unroll
    for (int r = 0; r < 8; r++) {
      Pw[(r + 8 * g) * 40 + mn]      = (__bf16)p0[r];
      Pw[(r + 8 * g) * 40 + 16 + mn] = (__bf16)p1[r];
    }
    __builtin_amdgcn_wave_barrier();
    {
      const __bf16* p = &Pw[mn * 40 + g * 8];
      v16bf ap = ld_frag(p, p + 16);
      // row sums of P via matrix math: P(16x32) x ones(32x16) broadcasts
      // each row-sum across all 16 lanes of the group (replaces shfl chain)
      v8f zc = {};
      v8f accRS = WMMA_BF16(ap, onesf, zc);
#pragma unroll
      for (int oc = 0; oc < 4; oc++) {
        const __bf16* q = &VtT[(oc * 16 + mn) * 40 + g * 8];
        v16bf bv = ld_frag(q, q + 16);
        accO[oc] = WMMA_BF16(ap, bv, accO[oc]);
      }
#pragma unroll
      for (int r = 0; r < 8; r++) srow[r] = srow[r] * corr[r] + accRS[r];
    }
    __syncthreads();
  }

#pragma unroll
  for (int oc = 0; oc < 4; oc++)
#pragma unroll
    for (int r = 0; r < 8; r++) {
      int row = q0 + w * 16 + r + 8 * g;
      int col = h * 64 + oc * 16 + mn;
      o[((size_t)b * S + row) * D + col] = (__bf16)(accO[oc][r] / srow[r]);
    }
}

// ---------------------------------------------------------------------------
// LayerNorm (+ optional exact-erf GELU), one row per block
// ---------------------------------------------------------------------------
template <bool GELU, bool OUTBF>
__global__ __launch_bounds__(256) void ln_act(const float* __restrict__ in,
                                              const float* __restrict__ gw,
                                              const float* __restrict__ bw,
                                              void* __restrict__ out, int W)
{
  __shared__ float r1[256], r2[256];
  const int row = blockIdx.x, tid = threadIdx.x;
  const float* x = in + (size_t)row * W;
  float s = 0.f, s2 = 0.f;
  for (int i = tid; i < W; i += 256) { float v = x[i]; s += v; s2 += v * v; }
  r1[tid] = s; r2[tid] = s2; __syncthreads();
  for (int st = 128; st > 0; st >>= 1) {
    if (tid < st) { r1[tid] += r1[tid + st]; r2[tid] += r2[tid + st]; }
    __syncthreads();
  }
  float mean = r1[0] / W;
  float var  = r2[0] / W - mean * mean;
  float inv  = rsqrtf(var + 1e-5f);
  for (int i = tid; i < W; i += 256) {
    float v = (x[i] - mean) * inv * gw[i] + bw[i];
    if (GELU) v = 0.5f * v * (1.f + erff(v * 0.70710678118654752f));
    if (OUTBF) ((__bf16*)out)[(size_t)row * W + i] = (__bf16)v;
    else       ((float*)out)[(size_t)row * W + i]  = v;
  }
}

// ---------------------------------------------------------------------------
// Vector-quantize: per-token argmin over 512 codebook rows (first-min tie-break),
// writes quantized row (== forward x_q) to d_out, bf16 copy, per-token SSE.
// ---------------------------------------------------------------------------
__global__ __launch_bounds__(256) void vq_kernel(const float* __restrict__ enc,
                                                 const float* __restrict__ cb,
                                                 float* __restrict__ xq,
                                                 __bf16* __restrict__ xqb,
                                                 float* __restrict__ partial)
{
  __shared__ float e[256];
  __shared__ float bv[256];
  __shared__ int   bi[256];
  const int t = blockIdx.x, tid = threadIdx.x;
  e[tid] = enc[(size_t)t * 256 + tid];
  __syncthreads();
  float bestv = 3.0e38f; int besti = 0;
  for (int c = tid; c < 512; c += 256) {
    const float* cr = cb + (size_t)c * 256;
    float d = 0.f;
    for (int i = 0; i < 256; i++) { float df = e[i] - cr[i]; d += df * df; }
    if (d < bestv) { bestv = d; besti = c; }
  }
  bv[tid] = bestv; bi[tid] = besti; __syncthreads();
  for (int st = 128; st > 0; st >>= 1) {
    if (tid < st) {
      if (bv[tid + st] < bv[tid] || (bv[tid + st] == bv[tid] && bi[tid + st] < bi[tid])) {
        bv[tid] = bv[tid + st]; bi[tid] = bi[tid + st];
      }
    }
    __syncthreads();
  }
  const int best = bi[0];
  __syncthreads();
  float q = cb[(size_t)best * 256 + tid];
  xq[(size_t)t * 256 + tid]  = q;
  xqb[(size_t)t * 256 + tid] = (__bf16)q;
  float df = e[tid] - q;
  bv[tid] = df * df; __syncthreads();
  for (int st = 128; st > 0; st >>= 1) {
    if (tid < st) bv[tid] += bv[tid + st];
    __syncthreads();
  }
  if (tid == 0) partial[t] = bv[0];
}

__global__ __launch_bounds__(256) void commit_reduce(const float* __restrict__ partial,
                                                     float* __restrict__ loss, int n, float scale)
{
  __shared__ float red[256];
  float s = 0.f;
  for (int i = threadIdx.x; i < n; i += 256) s += partial[i];
  red[threadIdx.x] = s; __syncthreads();
  for (int st = 128; st > 0; st >>= 1) {
    if (threadIdx.x < st) red[threadIdx.x] += red[threadIdx.x + st];
    __syncthreads();
  }
  if (threadIdx.x == 0) loss[0] = red[0] * scale;
}

// ---------------------------------------------------------------------------
// 128->1 head (dot + bias, optional sigmoid), one token per block
// ---------------------------------------------------------------------------
__global__ __launch_bounds__(128) void head_kernel(const __bf16* __restrict__ hbuf,
                                                   const float* __restrict__ w2,
                                                   const float* __restrict__ b2,
                                                   float* __restrict__ out, int sig)
{
  __shared__ float red[128];
  const int t = blockIdx.x, tid = threadIdx.x;
  red[tid] = (float)hbuf[(size_t)t * 128 + tid] * w2[tid];
  __syncthreads();
  for (int st = 64; st > 0; st >>= 1) {
    if (tid < st) red[tid] += red[tid + st];
    __syncthreads();
  }
  if (tid == 0) {
    float r = red[0] + b2[0];
    if (sig) r = 1.f / (1.f + __expf(-r));
    out[t] = r;
  }
}

__global__ void concat_k(const __bf16* __restrict__ xb, const __bf16* __restrict__ xqb,
                         __bf16* __restrict__ cat, size_t n)
{
  size_t i  = (size_t)blockIdx.x * blockDim.x + threadIdx.x;
  size_t st = (size_t)gridDim.x * blockDim.x;
  for (; i < n; i += st) {
    size_t t = i / 1280, c = i % 1280;
    cat[i] = (c < 1024) ? xb[t * 1024 + c] : xqb[t * 256 + (c - 1024)];
  }
}

__global__ void final_mix(const float* __restrict__ gz, const float* __restrict__ dec,
                          const float* __restrict__ x, float* __restrict__ out, size_t n)
{
  size_t i  = (size_t)blockIdx.x * blockDim.x + threadIdx.x;
  size_t st = (size_t)gridDim.x * blockDim.x;
  for (; i < n; i += st) {
    float g = 1.f / (1.f + __expf(-gz[i]));
    out[i] = g * dec[i] + (1.f - g) * x[i];
  }
}

// ---------------------------------------------------------------------------
// Host orchestration
// ---------------------------------------------------------------------------
extern "C" void kernel_launch(void* const* d_in, const int* in_sizes, int n_in,
                              void* d_out, int out_size, void* d_ws, size_t ws_size,
                              hipStream_t stream)
{
  (void)in_sizes; (void)n_in; (void)out_size; (void)ws_size;
  const int Dm = 1024, Nt = 8192;  // N tokens = B*S

  const float* x        = (const float*)d_in[0];
  const float* codebook = (const float*)d_in[1];
  const float* in_w  = (const float*)d_in[2];
  const float* in_b  = (const float*)d_in[3];
  const float* out_w = (const float*)d_in[4];
  const float* out_b = (const float*)d_in[5];
  const float* se_w1 = (const float*)d_in[6];
  const float* se_b1 = (const float*)d_in[7];
  const float* se_g1 = (const float*)d_in[8];
  const float* se_be1= (const float*)d_in[9];
  const float* se_w2 = (const float*)d_in[10];
  const float* se_b2 = (const float*)d_in[11];
  const float* en_w1 = (const float*)d_in[12];
  const float* en_b1 = (const float*)d_in[13];
  const float* en_g1 = (const float*)d_in[14];
  const float* en_be1= (const float*)d_in[15];
  const float* en_w2 = (const float*)d_in[16];
  const float* en_b2 = (const float*)d_in[17];
  const float* en_g2 = (const float*)d_in[18];
  const float* en_be2= (const float*)d_in[19];
  const float* ga_w  = (const float*)d_in[20];
  const float* ga_b  = (const float*)d_in[21];
  const float* de_w1 = (const float*)d_in[22];
  const float* de_b1 = (const float*)d_in[23];
  const float* de_g1 = (const float*)d_in[24];
  const float* de_be1= (const float*)d_in[25];
  const float* de_w2 = (const float*)d_in[26];
  const float* de_b2 = (const float*)d_in[27];
  const float* de_g2 = (const float*)d_in[28];
  const float* de_be2= (const float*)d_in[29];
  const float* co_w1 = (const float*)d_in[30];
  const float* co_b1 = (const float*)d_in[31];
  const float* co_g1 = (const float*)d_in[32];
  const float* co_be1= (const float*)d_in[33];
  const float* co_w2 = (const float*)d_in[34];
  const float* co_b2 = (const float*)d_in[35];
  const float* et_w1 = (const float*)d_in[36];
  const float* et_b1 = (const float*)d_in[37];
  const float* et_g1 = (const float*)d_in[38];
  const float* et_be1= (const float*)d_in[39];
  const float* et_w2 = (const float*)d_in[40];
  const float* et_b2 = (const float*)d_in[41];

  // d_out layout (tuple flattened in return order)
  float* out_f    = (float*)d_out;
  float* xq_out   = out_f;                 // [8,1024,256]
  float* rec_out  = out_f + 2097152;       // [8,1024,1024]
  float* loss_out = out_f + 10485760;      // scalar
  float* cons_out = out_f + 10485761;      // [8,1024,1]
  float* ent_out  = out_f + 10493953;      // [8,1024,1]
  float* ac_out   = out_f + 10502145;      // [8,1024,256]

  // workspace carve-out
  char* wsp = (char*)d_ws;
  size_t off = 0;
  auto alloc = [&](size_t bytes) -> char* {
    char* p = wsp + off;
    off = (off + bytes + 255) & ~(size_t)255;
    return p;
  };
  __bf16* xb    = (__bf16*)alloc((size_t)Nt * Dm * 2);
  __bf16* qkvb  = (__bf16*)alloc((size_t)Nt * 3 * Dm * 2);
  __bf16* attnb = (__bf16*)alloc((size_t)Nt * Dm * 2);
  float*  f32b  = (float*) alloc((size_t)Nt * 2048 * 4);
  __bf16* bfb   = (__bf16*)alloc((size_t)Nt * 2048 * 2);
  float*  enc32 = (float*) alloc((size_t)Nt * 256 * 4);
  __bf16* xqb   = (__bf16*)alloc((size_t)Nt * 256 * 2);
  __bf16* catb  = (__bf16*)alloc((size_t)Nt * 1280 * 2);
  float*  dec32 = (float*) alloc((size_t)Nt * Dm * 4);
  float*  part  = (float*) alloc((size_t)Nt * 4);
  __bf16* wb_in  = (__bf16*)alloc(3072ull * 1024 * 2);
  __bf16* wb_out = (__bf16*)alloc(1024ull * 1024 * 2);
  __bf16* wb_se1 = (__bf16*)alloc(2048ull * 1024 * 2);
  __bf16* wb_se2 = (__bf16*)alloc(256ull  * 2048 * 2);
  __bf16* wb_en1 = (__bf16*)alloc(2048ull * 1024 * 2);
  __bf16* wb_en2 = (__bf16*)alloc(256ull  * 2048 * 2);
  __bf16* wb_ga  = (__bf16*)alloc(1024ull * 1280 * 2);
  __bf16* wb_de1 = (__bf16*)alloc(2048ull * 256  * 2);
  __bf16* wb_de2 = (__bf16*)alloc(1024ull * 2048 * 2);
  __bf16* wb_co1 = (__bf16*)alloc(128ull  * 256  * 2);
  __bf16* wb_et1 = (__bf16*)alloc(128ull  * 256  * 2);

  auto cvt = [&](const float* s, __bf16* d, size_t n) {
    int blocks = (int)((n + 2047) / 2048);
    if (blocks > 4096) blocks = 4096;
    cvt_f32_bf16<<<blocks, 256, 0, stream>>>(s, d, n);
  };
  auto gemm_bf = [&](const __bf16* A, const __bf16* W, const float* bias, __bf16* C,
                     int M, int N, int K) {
    gemm_bf16_wmma<true><<<dim3(N / 128, M / 128), 256, 0, stream>>>(A, W, bias, C, M, N, K);
  };
  auto gemm_f = [&](const __bf16* A, const __bf16* W, const float* bias, float* C,
                    int M, int N, int K) {
    gemm_bf16_wmma<false><<<dim3(N / 128, M / 128), 256, 0, stream>>>(A, W, bias, C, M, N, K);
  };

  // weight/activation bf16 conversion
  cvt(x, xb, (size_t)Nt * Dm);
  cvt(in_w,  wb_in,  3072ull * 1024);
  cvt(out_w, wb_out, 1024ull * 1024);
  cvt(se_w1, wb_se1, 2048ull * 1024);
  cvt(se_w2, wb_se2, 256ull  * 2048);
  cvt(en_w1, wb_en1, 2048ull * 1024);
  cvt(en_w2, wb_en2, 256ull  * 2048);
  cvt(ga_w,  wb_ga,  1024ull * 1280);
  cvt(de_w1, wb_de1, 2048ull * 256);
  cvt(de_w2, wb_de2, 1024ull * 2048);
  cvt(co_w1, wb_co1, 128ull  * 256);
  cvt(et_w1, wb_et1, 128ull  * 256);

  // --- attention ---
  gemm_bf(xb, wb_in, in_b, qkvb, Nt, 3072, 1024);
  flash_attn<<<dim3(16, 16, 8), 128, 0, stream>>>(qkvb, attnb);
  gemm_f(attnb, wb_out, out_b, f32b, Nt, 1024, 1024);
  cvt(f32b, bfb, (size_t)Nt * 1024);  // attended -> bf16

  // --- structure encoder ---
  gemm_f(bfb, wb_se1, se_b1, f32b, Nt, 2048, 1024);
  ln_act<true, true><<<Nt, 256, 0, stream>>>(f32b, se_g1, se_be1, bfb, 2048);
  gemm_f(bfb, wb_se2, se_b2, ac_out, Nt, 256, 2048);

  // --- encoder ---
  gemm_f(xb, wb_en1, en_b1, f32b, Nt, 2048, 1024);
  ln_act<true, true><<<Nt, 256, 0, stream>>>(f32b, en_g1, en_be1, bfb, 2048);
  gemm_f(bfb, wb_en2, en_b2, f32b, Nt, 256, 2048);
  ln_act<false, false><<<Nt, 256, 0, stream>>>(f32b, en_g2, en_be2, enc32, 256);

  // --- vector quantize + commitment loss ---
  vq_kernel<<<Nt, 256, 0, stream>>>(enc32, codebook, xq_out, xqb, part);
  commit_reduce<<<1, 256, 0, stream>>>(part, loss_out, Nt, 1.f / ((float)Nt * 256.f));

  // --- consistency head ---
  gemm_f(xqb, wb_co1, co_b1, f32b, Nt, 128, 256);
  ln_act<true, true><<<Nt, 256, 0, stream>>>(f32b, co_g1, co_be1, bfb, 128);
  head_kernel<<<Nt, 128, 0, stream>>>(bfb, co_w2, co_b2, cons_out, 1);

  // --- entropy head ---
  gemm_f(xqb, wb_et1, et_b1, f32b, Nt, 128, 256);
  ln_act<true, true><<<Nt, 256, 0, stream>>>(f32b, et_g1, et_be1, bfb, 128);
  head_kernel<<<Nt, 128, 0, stream>>>(bfb, et_w2, et_b2, ent_out, 0);

  // --- decoder ---
  gemm_f(xqb, wb_de1, de_b1, f32b, Nt, 2048, 256);
  ln_act<true, true><<<Nt, 256, 0, stream>>>(f32b, de_g1, de_be1, bfb, 2048);
  gemm_f(bfb, wb_de2, de_b2, f32b, Nt, 1024, 2048);
  ln_act<false, false><<<Nt, 256, 0, stream>>>(f32b, de_g2, de_be2, dec32, 1024);

  // --- gate + residual mix ---
  concat_k<<<2048, 256, 0, stream>>>(xb, xqb, catb, (size_t)Nt * 1280);
  gemm_f(catb, wb_ga, ga_b, f32b, Nt, 1024, 1280);
  final_mix<<<2048, 256, 0, stream>>>(f32b, dec32, x, rec_out, (size_t)Nt * Dm);
}
